// PTA_EGNN_62122406969670
// MI455X (gfx1250) — matrix operand order repaired and checked by reading, not compiled, not run on hardware
//
#include <hip/hip_runtime.h>
#include <hip/hip_bf16.h>

// ---------------------------------------------------------------------------
// PTA-EGNN forward for gfx1250 (MI455X).  All edge/node MLP GEMMs run on
// v_wmma_f32_16x16x32_f16 (16-row tiles; E and N are multiples of 16).
// Segment ops (mean / softmax) use fp32 atomics + order-preserving uint max.
// ---------------------------------------------------------------------------

typedef __attribute__((ext_vector_type(16))) _Float16 v16h;
typedef __attribute__((ext_vector_type(8)))  _Float16 v8h;
typedef __attribute__((ext_vector_type(8)))  float    v8f;

#define DEVINL __device__ __forceinline__

constexpr int EN = 320000;   // edges
constexpr int NN = 20000;    // nodes

constexpr size_t al(size_t x){ return (x + 255) & ~(size_t)255; }

// ---- workspace layout (bytes) ----
constexpr size_t O_XCHEM = 0;                                   // [E,160] f16
constexpr size_t O_XPOS  = al(O_XCHEM + (size_t)EN*160*2);      // [E,64]  f16
constexpr size_t O_UNIT  = al(O_XPOS  + (size_t)EN*64*2);       // [E,3]   f32
constexpr size_t O_B1    = al(O_UNIT  + (size_t)EN*3*4);        // chem [E,64] f16
constexpr size_t O_B2    = al(O_B1    + (size_t)EN*64*2);       // pos  [E,64] f16
constexpr size_t O_B3    = al(O_B2    + (size_t)EN*64*2);       // scratch f16
constexpr size_t O_B4    = al(O_B3    + (size_t)EN*64*2);       // z / scratch f16
constexpr size_t O_ATT   = al(O_B4    + (size_t)EN*64*2);       // [E,4] f32
constexpr size_t O_PHIU  = al(O_ATT   + (size_t)EN*4*4);        // [E] f32
constexpr size_t O_PHIX  = al(O_PHIU  + (size_t)EN*4);          // [E] f32
constexpr size_t O_NF    = al(O_PHIX  + (size_t)EN*4);          // [N,9] f32
constexpr size_t O_DEG   = al(O_NF    + (size_t)NN*9*4);        // [N] f32
constexpr size_t O_NVEC  = al(O_DEG   + (size_t)NN*4);          // [N,9] f32
constexpr size_t O_AMAX  = al(O_NVEC  + (size_t)NN*9*4);        // [N,4] u32 keys
constexpr size_t O_SSUM  = al(O_AMAX  + (size_t)NN*4*4);        // [N,4] f32
constexpr size_t O_MAGG  = al(O_SSUM  + (size_t)NN*4*4);        // [N,64] f32
constexpr size_t O_XAGG  = al(O_MAGG  + (size_t)NN*64*4);       // [N,3] f32
constexpr size_t O_XNODE = al(O_XAGG  + (size_t)NN*3*4);        // [N,160] f16
constexpr size_t O_GBUF  = al(O_XNODE + (size_t)NN*160*2);      // [4,64,64] f32
// B-fragment weight packs (f16).  KP=160 pack = 4*5*512 halfs; KP=64 = 4*2*512.
constexpr size_t SZP160  = (size_t)4*5*512*2;
constexpr size_t SZP64   = (size_t)4*2*512*2;
constexpr size_t O_CHW1P = al(O_GBUF  + (size_t)4*64*64*4);
constexpr size_t O_PHW1P = al(O_CHW1P + SZP160);
constexpr size_t O_POSW1P= al(O_PHW1P + SZP160);
constexpr size_t O_CHW2P = al(O_POSW1P+ SZP64);
constexpr size_t O_POSW2P= al(O_CHW2P + SZP64);
constexpr size_t O_SHW1P = al(O_POSW2P+ SZP64);
constexpr size_t O_SHW2P = al(O_SHW1P + SZP64);
constexpr size_t O_TOW1P = al(O_SHW2P + SZP64);
constexpr size_t O_TOW2P = al(O_TOW1P + SZP64);
constexpr size_t O_PUW1P = al(O_TOW2P + SZP64);
constexpr size_t O_PXW1P = al(O_PUW1P + SZP64);
constexpr size_t O_PHW2P = al(O_PXW1P + SZP64);
constexpr size_t O_GP    = al(O_PHW2P + SZP64);   // 4 heads
constexpr size_t O_WVP   = al(O_GP    + 4*SZP64); // 4 heads

// ---------------------------------------------------------------------------
DEVINL v8f wmma16(v16h a, v16h b, v8f c){
  return __builtin_amdgcn_wmma_f32_16x16x32_f16(false, a, false, b,
                                                (short)0, c, false, false);
}

DEVINL v16h load_a_frag(const _Float16* __restrict__ X, int KP, int e0, int kb,
                        int lane){
  int mrow = lane & 15, half = lane >> 4;
  const _Float16* p = X + (size_t)(e0 + mrow)*KP + kb*32 + half*8;
  v8h a0 = *(const v8h*)p;
  v8h a1 = *(const v8h*)(p + 16);
  return __builtin_shufflevector(a0, a1, 0,1,2,3,4,5,6,7,8,9,10,11,12,13,14,15);
}

DEVINL v16h load_b_frag(const _Float16* __restrict__ Wp, int KB, int t, int kb,
                        int lane){
  return *(const v16h*)(Wp + (((size_t)t*KB + kb)*32 + lane)*16);
}

// order-preserving float<->uint key for atomic segment max
DEVINL unsigned fenc(float f){
  unsigned u = __float_as_uint(f);
  return (u & 0x80000000u) ? ~u : (u | 0x80000000u);
}
DEVINL float fdec(unsigned k){
  unsigned u = (k & 0x80000000u) ? (k & 0x7fffffffu) : ~k;
  return __uint_as_float(u);
}

// ---------------------------------------------------------------------------
// Weight packing into B-fragment order.
__global__ void k_pack(const float* __restrict__ W, _Float16* __restrict__ Wp,
                       int Kreal, int KB){
  int idx = blockIdx.x*blockDim.x + threadIdx.x;
  int total = 4*KB*512;
  if (idx >= total) return;
  int i  = idx & 15;
  int L  = (idx >> 4) & 31;
  int tk = idx >> 9;          // t*KB + kb
  int kb = tk % KB;
  int t  = tk / KB;
  int k  = kb*32 + ((i >> 3) << 4) + ((L >> 4) << 3) + (i & 7);
  int n  = (t << 4) + (L & 15);
  float v = (k < Kreal) ? W[k*64 + n] : 0.0f;
  Wp[idx] = (_Float16)v;
}

// G_h = Wq_h @ Wk_h^T  (per-head 64x64)
__global__ void k_G(const float* __restrict__ wq, const float* __restrict__ wk,
                    float* __restrict__ G){
  int idx = blockIdx.x*blockDim.x + threadIdx.x;
  if (idx >= 4*64*64) return;
  int h = idx >> 12, d = (idx >> 6) & 63, dp = idx & 63;
  const float* q = wq + h*4096 + d*64;
  const float* k = wk + h*4096 + dp*64;
  float s = 0.f;
  #pragma unroll 8
  for (int f = 0; f < 64; ++f) s += q[f]*k[f];
  G[idx] = s;
}

// ---------------------------------------------------------------------------
// Geometry per edge: radial basis, unit_diff, nforce scatter, degree.
__global__ void k_geom(const float* __restrict__ coord, const int* __restrict__ edges,
                       const float* __restrict__ eattr, float* __restrict__ nf,
                       float* __restrict__ deg, _Float16* __restrict__ Xpos,
                       float* __restrict__ unit){
  int e = blockIdx.x*blockDim.x + threadIdx.x;
  if (e >= EN) return;
  int r = edges[e], c = edges[EN + e];
  float d0 = coord[r*3+0] - coord[c*3+0];
  float d1 = coord[r*3+1] - coord[c*3+1];
  float d2 = coord[r*3+2] - coord[c*3+2];
  float r2 = d0*d0 + d1*d1 + d2*d2;
  float nrm = sqrtf(r2);
  float iu = 1.f/(nrm + 1e-8f);
  unit[(size_t)e*3+0] = d0*iu; unit[(size_t)e*3+1] = d1*iu; unit[(size_t)e*3+2] = d2*iu;
  float invn = 1.f/nrm;
  float p3 = invn*invn*invn, p4 = p3*invn, p5 = p4*invn;
  float pw[3] = {p3, p4, p5};
  float dd[3] = {d0, d1, d2};
  #pragma unroll
  for (int a = 0; a < 3; ++a)
    #pragma unroll
    for (int cc = 0; cc < 3; ++cc)
      atomicAdd(&nf[(size_t)r*9 + a*3 + cc], pw[a]*dd[cc]);
  atomicAdd(&deg[r], 1.f);
  _Float16* xp = Xpos + (size_t)e*64;
  float s = -0.5f;
  #pragma unroll
  for (int j = 0; j < 15; ++j){ xp[3+j] = (_Float16)__expf(r2*s); s *= (1.f/2.25f); }
  #pragma unroll
  for (int j = 0; j < 16; ++j) xp[18+j] = (_Float16)eattr[(size_t)e*16+j];
  #pragma unroll
  for (int j = 34; j < 64; ++j) xp[j] = (_Float16)0.f;
}

// Build Xchem[E,160] = [h[row], h[col], edge_attr, pad]
__global__ void k_xchem(const float* __restrict__ h, const int* __restrict__ edges,
                        const float* __restrict__ eattr, _Float16* __restrict__ X){
  int idx = blockIdx.x*blockDim.x + threadIdx.x;
  if (idx >= EN*160) return;
  int e = idx/160, k = idx - e*160;
  float v;
  if (k < 64)        v = h[(size_t)edges[e]*64 + k];
  else if (k < 128)  v = h[(size_t)edges[EN+e]*64 + (k-64)];
  else if (k < 144)  v = eattr[(size_t)e*16 + (k-128)];
  else               v = 0.f;
  X[idx] = (_Float16)v;
}

__global__ void k_nvec(const float* __restrict__ nf, const float* __restrict__ deg,
                       float* __restrict__ nv){
  int n = blockIdx.x*blockDim.x + threadIdx.x;
  if (n >= NN) return;
  float d = fmaxf(deg[n], 1.f);
  #pragma unroll
  for (int a = 0; a < 3; ++a){
    float v0 = nf[(size_t)n*9+a*3+0]/d;
    float v1 = nf[(size_t)n*9+a*3+1]/d;
    float v2 = nf[(size_t)n*9+a*3+2]/d;
    float nr = sqrtf(v0*v0+v1*v1+v2*v2) + 1e-8f;
    nv[(size_t)n*9+a*3+0] = v0/nr;
    nv[(size_t)n*9+a*3+1] = v1/nr;
    nv[(size_t)n*9+a*3+2] = v2/nr;
  }
}

__global__ void k_nprod(const float* __restrict__ nv, const int* __restrict__ edges,
                        _Float16* __restrict__ Xpos){
  int e = blockIdx.x*blockDim.x + threadIdx.x;
  if (e >= EN) return;
  int r = edges[e], c = edges[EN + e];
  #pragma unroll
  for (int a = 0; a < 3; ++a){
    float s = 0.f;
    #pragma unroll
    for (int cc = 0; cc < 3; ++cc)
      s += nv[(size_t)r*9+a*3+cc]*nv[(size_t)c*9+a*3+cc];
    Xpos[(size_t)e*64 + a] = (_Float16)s;
  }
}

// ---------------------------------------------------------------------------
// Generic 16-row WMMA GEMM: Y[16,64] = act(X[16,KP] @ W + b), with epilogues.
// MODE 0: SiLU -> f16   1: plain -> f16
// MODE 2: out=(XW+b)*pos; att=sigmoid(out.attw+attb); z=out*att -> f16
// MODE 3: m=XW+b; atomicAdd m_agg[row[e]]
// MODE 4: fout = 0.2*hsrc + 0.8*(XW+b)   (node feature update)
template<int KP, int MODE>
__global__ __launch_bounds__(256) void k_gemm(
    const _Float16* __restrict__ X, const _Float16* __restrict__ Wp,
    const float* __restrict__ bias, _Float16* __restrict__ Y,
    const _Float16* __restrict__ posbuf, const float* __restrict__ attw,
    const float* __restrict__ attb, const int* __restrict__ rowidx,
    float* __restrict__ magg, const float* __restrict__ hsrc,
    float* __restrict__ fout, int nrows){
  constexpr int KB = KP/32;
  int tile = blockIdx.x*(blockDim.x >> 5) + (threadIdx.x >> 5);
  int e0 = tile << 4;
  if (e0 >= nrows) return;
  int lane = threadIdx.x & 31;
  int half = lane >> 4;
  int nn   = lane & 15;
  v8f acc[4] = {};
  for (int kb = 0; kb < KB; ++kb){
    v16h a = load_a_frag(X, KP, e0, kb, lane);
    #pragma unroll
    for (int t = 0; t < 4; ++t){
      v16h b = load_b_frag(Wp, KB, t, kb, lane);
      acc[t] = wmma16(a, b, acc[t]);
    }
  }
  if (MODE == 0 || MODE == 1){
    #pragma unroll
    for (int t = 0; t < 4; ++t){
      float bv = bias[t*16 + nn];
      #pragma unroll
      for (int r = 0; r < 8; ++r){
        float v = acc[t][r] + bv;
        if (MODE == 0) v = v/(1.f + __expf(-v));
        Y[(size_t)(e0 + r + 8*half)*64 + t*16 + nn] = (_Float16)v;
      }
    }
  } else if (MODE == 2){
    float outv[4][8], part[8];
    #pragma unroll
    for (int r = 0; r < 8; ++r) part[r] = 0.f;
    #pragma unroll
    for (int t = 0; t < 4; ++t){
      float bv = bias[t*16 + nn];
      float wv = attw[t*16 + nn];
      #pragma unroll
      for (int r = 0; r < 8; ++r){
        float o = acc[t][r] + bv;
        o *= (float)posbuf[(size_t)(e0 + r + 8*half)*64 + t*16 + nn];
        outv[t][r] = o;
        part[r] += o*wv;
      }
    }
    #pragma unroll
    for (int off = 1; off < 16; off <<= 1)
      #pragma unroll
      for (int r = 0; r < 8; ++r) part[r] += __shfl_xor(part[r], off, 32);
    float ab = attb[0];
    #pragma unroll
    for (int t = 0; t < 4; ++t)
      #pragma unroll
      for (int r = 0; r < 8; ++r){
        float av = 1.f/(1.f + __expf(-(part[r] + ab)));
        Y[(size_t)(e0 + r + 8*half)*64 + t*16 + nn] = (_Float16)(outv[t][r]*av);
      }
  } else if (MODE == 3){
    #pragma unroll
    for (int t = 0; t < 4; ++t){
      float bv = bias[t*16 + nn];
      #pragma unroll
      for (int r = 0; r < 8; ++r){
        int e = e0 + r + 8*half;
        atomicAdd(&magg[(size_t)rowidx[e]*64 + t*16 + nn], acc[t][r] + bv);
      }
    }
  } else {
    #pragma unroll
    for (int t = 0; t < 4; ++t){
      float bv = bias[t*16 + nn];
      #pragma unroll
      for (int r = 0; r < 8; ++r){
        size_t o = (size_t)(e0 + r + 8*half)*64 + t*16 + nn;
        fout[o] = 0.2f*hsrc[o] + 0.8f*(acc[t][r] + bv);
      }
    }
  }
}

// att[e,h] = rowdot(z @ G_h, z) / 8
__global__ __launch_bounds__(256) void k_att(const _Float16* __restrict__ Z,
    const _Float16* __restrict__ Gp, float* __restrict__ att, int nrows){
  int tile = blockIdx.x*(blockDim.x >> 5) + (threadIdx.x >> 5);
  int e0 = tile << 4;
  if (e0 >= nrows) return;
  int lane = threadIdx.x & 31, half = lane >> 4, nn = lane & 15;
  v16h afr[2];
  #pragma unroll
  for (int kb = 0; kb < 2; ++kb) afr[kb] = load_a_frag(Z, 64, e0, kb, lane);
  float zc[4][8];
  #pragma unroll
  for (int t = 0; t < 4; ++t)
    #pragma unroll
    for (int r = 0; r < 8; ++r)
      zc[t][r] = (float)Z[(size_t)(e0 + r + 8*half)*64 + t*16 + nn];
  for (int h = 0; h < 4; ++h){
    const _Float16* gp = Gp + (size_t)h*4096;
    v8f acc[4] = {};
    #pragma unroll
    for (int kb = 0; kb < 2; ++kb)
      #pragma unroll
      for (int t = 0; t < 4; ++t)
        acc[t] = wmma16(afr[kb], load_b_frag(gp, 2, t, kb, lane), acc[t]);
    float part[8];
    #pragma unroll
    for (int r = 0; r < 8; ++r) part[r] = 0.f;
    #pragma unroll
    for (int t = 0; t < 4; ++t)
      #pragma unroll
      for (int r = 0; r < 8; ++r) part[r] += acc[t][r]*zc[t][r];
    #pragma unroll
    for (int off = 1; off < 16; off <<= 1)
      #pragma unroll
      for (int r = 0; r < 8; ++r) part[r] += __shfl_xor(part[r], off, 32);
    if (nn == 0){
      #pragma unroll
      for (int r = 0; r < 8; ++r)
        att[(size_t)(e0 + r + 8*half)*4 + h] = part[r]*0.125f;
    }
  }
}

// z_upd = sum_h a[:,h] * (z @ Wv_h)
__global__ __launch_bounds__(256) void k_zupd(const _Float16* __restrict__ Z,
    const _Float16* __restrict__ Wvp, const float* __restrict__ a,
    _Float16* __restrict__ out, int nrows){
  int tile = blockIdx.x*(blockDim.x >> 5) + (threadIdx.x >> 5);
  int e0 = tile << 4;
  if (e0 >= nrows) return;
  int lane = threadIdx.x & 31, half = lane >> 4, nn = lane & 15;
  v16h afr[2];
  #pragma unroll
  for (int kb = 0; kb < 2; ++kb) afr[kb] = load_a_frag(Z, 64, e0, kb, lane);
  v8f acc[4] = {};
  for (int h = 0; h < 4; ++h){
    const _Float16* wp = Wvp + (size_t)h*4096;
    v8f vac[4] = {};
    #pragma unroll
    for (int kb = 0; kb < 2; ++kb)
      #pragma unroll
      for (int t = 0; t < 4; ++t)
        vac[t] = wmma16(afr[kb], load_b_frag(wp, 2, t, kb, lane), vac[t]);
    float ar[8];
    #pragma unroll
    for (int r = 0; r < 8; ++r) ar[r] = a[(size_t)(e0 + r + 8*half)*4 + h];
    #pragma unroll
    for (int t = 0; t < 4; ++t)
      #pragma unroll
      for (int r = 0; r < 8; ++r) acc[t][r] += ar[r]*vac[t][r];
  }
  #pragma unroll
  for (int t = 0; t < 4; ++t)
    #pragma unroll
    for (int r = 0; r < 8; ++r)
      out[(size_t)(e0 + r + 8*half)*64 + t*16 + nn] = (_Float16)acc[t][r];
}

// segment softmax pieces
__global__ void k_segmax(const float* __restrict__ att, const int* __restrict__ edges,
                         unsigned* __restrict__ amax){
  int i = blockIdx.x*blockDim.x + threadIdx.x;
  if (i >= EN*4) return;
  int e = i >> 2, h = i & 3;
  atomicMax(&amax[(size_t)edges[e]*4 + h], fenc(att[i]));
}
__global__ void k_segexp(float* __restrict__ att, const int* __restrict__ edges,
                         const unsigned* __restrict__ amax, float* __restrict__ ssum){
  int i = blockIdx.x*blockDim.x + threadIdx.x;
  if (i >= EN*4) return;
  int e = i >> 2, h = i & 3;
  float ex = __expf(att[i] - fdec(amax[(size_t)edges[e]*4 + h]));
  att[i] = ex;
  atomicAdd(&ssum[(size_t)edges[e]*4 + h], ex);
}
__global__ void k_afinal(float* __restrict__ att, const int* __restrict__ edges,
                         const float* __restrict__ ssum){
  int i = blockIdx.x*blockDim.x + threadIdx.x;
  if (i >= EN*4) return;
  int e = i >> 2, h = i & 3;
  att[i] = att[i]/ssum[(size_t)edges[e]*4 + h];
}

// phi = H @ w2 + b2  (64 -> 1)
__global__ void k_phidot(const _Float16* __restrict__ H, const float* __restrict__ w2,
                         const float* __restrict__ b2, float* __restrict__ phi){
  int e = blockIdx.x*blockDim.x + threadIdx.x;
  if (e >= EN) return;
  float s = b2[0];
  #pragma unroll 8
  for (int j = 0; j < 64; ++j) s += (float)H[(size_t)e*64 + j]*w2[j];
  phi[e] = s;
}

__global__ void k_xagg(const float* __restrict__ phiu, const float* __restrict__ phix,
                       const float* __restrict__ unit, const int* __restrict__ edges,
                       float* __restrict__ xagg){
  int e = blockIdx.x*blockDim.x + threadIdx.x;
  if (e >= EN) return;
  float v = phiu[e]*phix[e];
  int r = edges[e];
  #pragma unroll
  for (int c = 0; c < 3; ++c)
    atomicAdd(&xagg[(size_t)r*3 + c], unit[(size_t)e*3 + c]*v);
}

__global__ void k_coordout(const float* __restrict__ coord, const float* __restrict__ init,
                           const float* __restrict__ xagg, const float* __restrict__ deg,
                           float* __restrict__ out){
  int n = blockIdx.x*blockDim.x + threadIdx.x;
  if (n >= NN) return;
  float d = fmaxf(deg[n], 1.f);
  #pragma unroll
  for (int c = 0; c < 3; ++c)
    out[(size_t)NN*64 + (size_t)n*3 + c] =
        0.2f*init[(size_t)n*3 + c] + 0.8f*coord[(size_t)n*3 + c] + xagg[(size_t)n*3 + c]/d;
}

// Xnode[N,160] = [h, node_attr, m_agg/deg, pad]
__global__ void k_xnode(const float* __restrict__ h, const float* __restrict__ nattr,
                        const float* __restrict__ magg, const float* __restrict__ deg,
                        _Float16* __restrict__ X){
  int idx = blockIdx.x*blockDim.x + threadIdx.x;
  if (idx >= NN*160) return;
  int n = idx/160, k = idx - n*160;
  float v;
  if (k < 64)       v = h[(size_t)n*64 + k];
  else if (k < 80)  v = nattr[(size_t)n*16 + (k-64)];
  else if (k < 144) v = magg[(size_t)n*64 + (k-80)]/fmaxf(deg[n], 1.f);
  else              v = 0.f;
  X[idx] = (_Float16)v;
}

// ---------------------------------------------------------------------------
extern "C" void kernel_launch(void* const* d_in, const int* in_sizes, int n_in,
                              void* d_out, int out_size, void* d_ws, size_t ws_size,
                              hipStream_t stream){
  (void)in_sizes; (void)n_in; (void)out_size; (void)ws_size;
  // input order: h, coord, edges, edge_attr, node_attr, init_coord, then params
  const float* h_in    = (const float*)d_in[0];
  const float* coord   = (const float*)d_in[1];
  const int*   edges   = (const int*)d_in[2];
  const float* eattr   = (const float*)d_in[3];
  const float* nattr   = (const float*)d_in[4];
  const float* initc   = (const float*)d_in[5];
  auto P = [&](int i){ return (const float*)d_in[i]; };
  // params leaves (dict order): 6 ch_w1,7 ch_b1,8 ch_w2,9 ch_b2,10 pos_w1,11 pos_b1,
  // 12 pos_w2,13 pos_b2,14 sh_w1,15 sh_b1,16 sh_w2,17 sh_b2,18 att_w,19 att_b,
  // 20 wq,21 wk,22 wv,23 to_w1,24 to_b1,25 to_w2,26 to_b2,27 pu_w1,28 pu_b1,
  // 29 pu_w2,30 pu_b2,31 px_w1,32 px_b1,33 px_w2,34 px_b2,35 ph_w1,36 ph_b1,
  // 37 ph_w2,38 ph_b2
  char* ws = (char*)d_ws;
  float* out = (float*)d_out;

  _Float16* XCHEM = (_Float16*)(ws + O_XCHEM);
  _Float16* XPOS  = (_Float16*)(ws + O_XPOS);
  float*    UNIT  = (float*)(ws + O_UNIT);
  _Float16* B1    = (_Float16*)(ws + O_B1);
  _Float16* B2    = (_Float16*)(ws + O_B2);
  _Float16* B3    = (_Float16*)(ws + O_B3);
  _Float16* B4    = (_Float16*)(ws + O_B4);
  float*    ATT   = (float*)(ws + O_ATT);
  float*    PHIU  = (float*)(ws + O_PHIU);
  float*    PHIX  = (float*)(ws + O_PHIX);
  float*    NF    = (float*)(ws + O_NF);
  float*    DEG   = (float*)(ws + O_DEG);
  float*    NVEC  = (float*)(ws + O_NVEC);
  unsigned* AMAX  = (unsigned*)(ws + O_AMAX);
  float*    SSUM  = (float*)(ws + O_SSUM);
  float*    MAGG  = (float*)(ws + O_MAGG);
  float*    XAGG  = (float*)(ws + O_XAGG);
  _Float16* XNODE = (_Float16*)(ws + O_XNODE);
  float*    GBUF  = (float*)(ws + O_GBUF);
  _Float16* CHW1P = (_Float16*)(ws + O_CHW1P);
  _Float16* PHW1P = (_Float16*)(ws + O_PHW1P);
  _Float16* POSW1P= (_Float16*)(ws + O_POSW1P);
  _Float16* CHW2P = (_Float16*)(ws + O_CHW2P);
  _Float16* POSW2P= (_Float16*)(ws + O_POSW2P);
  _Float16* SHW1P = (_Float16*)(ws + O_SHW1P);
  _Float16* SHW2P = (_Float16*)(ws + O_SHW2P);
  _Float16* TOW1P = (_Float16*)(ws + O_TOW1P);
  _Float16* TOW2P = (_Float16*)(ws + O_TOW2P);
  _Float16* PUW1P = (_Float16*)(ws + O_PUW1P);
  _Float16* PXW1P = (_Float16*)(ws + O_PXW1P);
  _Float16* PHW2P = (_Float16*)(ws + O_PHW2P);
  _Float16* GP    = (_Float16*)(ws + O_GP);
  _Float16* WVP   = (_Float16*)(ws + O_WVP);

  // zero accumulators
  hipMemsetAsync(NF,   0, (size_t)NN*9*4,  stream);
  hipMemsetAsync(DEG,  0, (size_t)NN*4,    stream);
  hipMemsetAsync(AMAX, 0, (size_t)NN*4*4,  stream);
  hipMemsetAsync(SSUM, 0, (size_t)NN*4*4,  stream);
  hipMemsetAsync(MAGG, 0, (size_t)NN*64*4, stream);
  hipMemsetAsync(XAGG, 0, (size_t)NN*3*4,  stream);

  // weight packing
  const int T = 256;
  int pt160 = 4*5*512, pt64 = 4*2*512;
  k_pack<<<(pt160+T-1)/T, T, 0, stream>>>(P(6),  CHW1P, 144, 5);
  k_pack<<<(pt160+T-1)/T, T, 0, stream>>>(P(35), PHW1P, 144, 5);
  k_pack<<<(pt64+T-1)/T,  T, 0, stream>>>(P(10), POSW1P, 34, 2);
  k_pack<<<(pt64+T-1)/T,  T, 0, stream>>>(P(8),  CHW2P,  64, 2);
  k_pack<<<(pt64+T-1)/T,  T, 0, stream>>>(P(12), POSW2P, 64, 2);
  k_pack<<<(pt64+T-1)/T,  T, 0, stream>>>(P(14), SHW1P,  64, 2);
  k_pack<<<(pt64+T-1)/T,  T, 0, stream>>>(P(16), SHW2P,  64, 2);
  k_pack<<<(pt64+T-1)/T,  T, 0, stream>>>(P(23), TOW1P,  64, 2);
  k_pack<<<(pt64+T-1)/T,  T, 0, stream>>>(P(25), TOW2P,  64, 2);
  k_pack<<<(pt64+T-1)/T,  T, 0, stream>>>(P(27), PUW1P,  64, 2);
  k_pack<<<(pt64+T-1)/T,  T, 0, stream>>>(P(31), PXW1P,  64, 2);
  k_pack<<<(pt64+T-1)/T,  T, 0, stream>>>(P(37), PHW2P,  64, 2);
  k_G<<<(16384+T-1)/T, T, 0, stream>>>(P(20), P(21), GBUF);
  for (int hh = 0; hh < 4; ++hh){
    k_pack<<<(pt64+T-1)/T, T, 0, stream>>>(GBUF + (size_t)hh*4096, GP + (size_t)hh*4096, 64, 2);
    k_pack<<<(pt64+T-1)/T, T, 0, stream>>>(P(22) + (size_t)hh*4096, WVP + (size_t)hh*4096, 64, 2);
  }

  // geometry + feature staging
  k_geom<<<(EN+T-1)/T, T, 0, stream>>>(coord, edges, eattr, NF, DEG, XPOS, UNIT);
  k_xchem<<<(EN*160+T-1)/T, T, 0, stream>>>(h_in, edges, eattr, XCHEM);
  k_nvec<<<(NN+T-1)/T, T, 0, stream>>>(NF, DEG, NVEC);
  k_nprod<<<(EN+T-1)/T, T, 0, stream>>>(NVEC, edges, XPOS);

  int eblk = (EN/16 + 7)/8;   // 8 waves per 256-thread block
  int nblk = (NN/16 + 7)/8;
  // chem = MLP2(Xchem)
  k_gemm<160,0><<<eblk,T,0,stream>>>(XCHEM, CHW1P, P(7), B3, nullptr,nullptr,nullptr,nullptr,nullptr,nullptr,nullptr, EN);
  k_gemm<64,1><<<eblk,T,0,stream>>>(B3, CHW2P, P(9), B1, nullptr,nullptr,nullptr,nullptr,nullptr,nullptr,nullptr, EN);
  // pos = MLP2(Xpos)
  k_gemm<64,0><<<eblk,T,0,stream>>>(XPOS, POSW1P, P(11), B3, nullptr,nullptr,nullptr,nullptr,nullptr,nullptr,nullptr, EN);
  k_gemm<64,1><<<eblk,T,0,stream>>>(B3, POSW2P, P(13), B2, nullptr,nullptr,nullptr,nullptr,nullptr,nullptr,nullptr, EN);
  // out = MLP2(chem)*pos ; z = out*sigmoid(out.attw+attb)
  k_gemm<64,0><<<eblk,T,0,stream>>>(B1, SHW1P, P(15), B3, nullptr,nullptr,nullptr,nullptr,nullptr,nullptr,nullptr, EN);
  k_gemm<64,2><<<eblk,T,0,stream>>>(B3, SHW2P, P(17), B4, B2, P(18), P(19), nullptr,nullptr,nullptr,nullptr, EN);
  // attention scores + segment softmax
  k_att<<<eblk,T,0,stream>>>(B4, GP, ATT, EN);
  k_segmax<<<(EN*4+T-1)/T, T, 0, stream>>>(ATT, edges, AMAX);
  k_segexp<<<(EN*4+T-1)/T, T, 0, stream>>>(ATT, edges, AMAX, SSUM);
  k_afinal<<<(EN*4+T-1)/T, T, 0, stream>>>(ATT, edges, SSUM);
  // z_upd and m; scatter m_agg
  k_zupd<<<eblk,T,0,stream>>>(B4, WVP, ATT, B3, EN);
  k_gemm<64,0><<<eblk,T,0,stream>>>(B3, TOW1P, P(24), B4, nullptr,nullptr,nullptr,nullptr,nullptr,nullptr,nullptr, EN);
  k_gemm<64,3><<<eblk,T,0,stream>>>(B4, TOW2P, P(26), nullptr, nullptr,nullptr,nullptr, edges, MAGG, nullptr,nullptr, EN);
  // coordinate gate phis
  k_gemm<64,0><<<eblk,T,0,stream>>>(B1, PUW1P, P(28), B3, nullptr,nullptr,nullptr,nullptr,nullptr,nullptr,nullptr, EN);
  k_phidot<<<(EN+T-1)/T, T, 0, stream>>>(B3, P(29), P(30), PHIU);
  k_gemm<64,0><<<eblk,T,0,stream>>>(B2, PXW1P, P(32), B3, nullptr,nullptr,nullptr,nullptr,nullptr,nullptr,nullptr, EN);
  k_phidot<<<(EN+T-1)/T, T, 0, stream>>>(B3, P(33), P(34), PHIX);
  k_xagg<<<(EN+T-1)/T, T, 0, stream>>>(PHIU, PHIX, UNIT, edges, XAGG);
  k_coordout<<<(NN+T-1)/T, T, 0, stream>>>(coord, initc, XAGG, DEG, out);
  // node feature update
  k_xnode<<<(NN*160+T-1)/T, T, 0, stream>>>(h_in, nattr, MAGG, DEG, XNODE);
  k_gemm<160,0><<<nblk,T,0,stream>>>(XNODE, PHW1P, P(36), B3, nullptr,nullptr,nullptr,nullptr,nullptr,nullptr,nullptr, NN);
  k_gemm<64,4><<<nblk,T,0,stream>>>(B3, PHW2P, P(38), nullptr, nullptr,nullptr,nullptr,nullptr,nullptr, h_in, out, NN);
}
